// RD_NOISING_7696581394521
// MI455X (gfx1250) — compile-verified, matrix operand order checked by reading, not run on hardware
//
#include <hip/hip_runtime.h>
#include <hip/hip_bf16.h>

typedef __attribute__((ext_vector_type(16))) _Float16 v16h;
typedef __attribute__((ext_vector_type(8)))  float    v8f;

#define N_PTS   100000
#define DIMS    256
#define NCORE   1000
#define TILES   ((N_PTS + 15) / 16)              // 6250 (exact: 6250*16 = 100000)
#define WPB     8                                 // waves (tiles) per block
#define BLKT    256
#define NBLK_A  ((TILES + WPB - 1) / WPB)         // 782

// ---------------------------------------------------------------------------
// Pass 0a: per-point squared norms + min_d = +inf.  One wave32 per point.
// ---------------------------------------------------------------------------
__global__ __launch_bounds__(BLKT) void fps_init_norms(
    const float* __restrict__ feat, float* __restrict__ nrm,
    float* __restrict__ min_d)
{
    int gid  = blockIdx.x * blockDim.x + threadIdx.x;
    int pt   = gid >> 5;
    int lane = gid & 31;
    if (pt >= N_PTS) return;
    const float* row = feat + (size_t)pt * DIMS;
    float s = 0.0f;
    #pragma unroll
    for (int j = 0; j < DIMS / 32; ++j) {         // coalesced: lane + j*32
        float x = row[lane + j * 32];
        s = fmaf(x, x, s);
    }
    #pragma unroll
    for (int off = 16; off > 0; off >>= 1)
        s += __shfl_xor(s, off, 32);
    if (lane == 0) {
        nrm[pt]   = s;
        min_d[pt] = __builtin_inff();
    }
}

// ---------------------------------------------------------------------------
// Pass 0b: seed = index 0.  Emit out row 0, center, center-norm.
// ---------------------------------------------------------------------------
__global__ __launch_bounds__(BLKT) void fps_init_seed(
    const float* __restrict__ feat, const float* __restrict__ nrm,
    float* __restrict__ out, float* __restrict__ center,
    float* __restrict__ cnrm)
{
    int t = threadIdx.x;                          // 0..255 == DIMS
    float v = feat[t];                            // row 0
    out[t]    = v;
    center[t] = v;
    if (t == 0) *cnrm = nrm[0];
}

// ---------------------------------------------------------------------------
// Step kernel A: WMMA distance sweep + min-update + block argmax partials.
// One wave per 16-point tile; dots via v_wmma_f32_16x16x32_f16 with the
// center replicated across all 16 B-columns.
// ---------------------------------------------------------------------------
__global__ __launch_bounds__(BLKT) void fps_step_dist(
    const float* __restrict__ feat, const float* __restrict__ center,
    const float* __restrict__ cnrm_p, const float* __restrict__ nrm,
    float* __restrict__ min_d, float* __restrict__ pval,
    int* __restrict__ pidx)
{
    __shared__ _Float16 sc[DIMS];
    __shared__ float    svals[BLKT];
    __shared__ int      sidxs[BLKT];

    int tid = threadIdx.x;
    sc[tid] = (_Float16)center[tid];              // stage f16 center in LDS
    __syncthreads();

    float cn   = *cnrm_p;
    int   wave = tid >> 5;
    int   lane = tid & 31;
    int   tile = blockIdx.x * WPB + wave;

    float best = -__builtin_inff();
    int   bidx = 0;

    if (tile < TILES) {                           // wave-uniform branch
        int row = lane & 15;                      // A: lanes r and r+16 share row r
        int khi = (lane >= 16) ? 8 : 0;           // K split per ISA layout
        const float* rbase = feat + ((size_t)tile * 16 + row) * DIMS + khi;

        v8f acc = {};
        #pragma unroll
        for (int ch = 0; ch < DIMS / 32; ++ch) {  // 8 chunks of K=32
            int cb = ch * 32;
            v16h a, b;
            const float*    fp = rbase + cb;      // K = cb+khi+0..7 and +16..23
            const _Float16* cp = &sc[cb + khi];
            #pragma unroll
            for (int j = 0; j < 8; ++j) {
                a[j]     = (_Float16)fp[j];
                a[8 + j] = (_Float16)fp[16 + j];
                b[j]     = cp[j];
                b[8 + j] = cp[16 + j];
            }
            acc = __builtin_amdgcn_wmma_f32_16x16x32_f16(
                false, a, false, b, (short)0, acc, false, false);
        }

        // C layout: lane<16 vgpr j = dot(row j); lane>=16 vgpr j = dot(row 8+j).
        // Lanes 0-7 and 16-23 each finalize one point; 8-15/24-31 are duplicates.
        if ((lane & 8) == 0) {
            int   sub = lane & 7;
            float dot = acc[0];
            #pragma unroll
            for (int j = 1; j < 8; ++j)
                if (sub == j) dot = acc[j];
            int p = tile * 16 + ((lane >= 16) ? 8 : 0) + sub;
            float nd = nrm[p] + cn - 2.0f * dot;  // squared distance (sqrt monotone)
            float nm = fminf(min_d[p], nd);
            min_d[p] = nm;
            best = nm;
            bidx = p;
        }
    }

    // Block argmax, first-occurrence tie rule (value desc, index asc).
    svals[tid] = best;
    sidxs[tid] = bidx;
    __syncthreads();
    #pragma unroll
    for (int s = BLKT / 2; s > 0; s >>= 1) {
        if (tid < s) {
            float v2 = svals[tid + s];
            int   i2 = sidxs[tid + s];
            if (v2 > svals[tid] || (v2 == svals[tid] && i2 < sidxs[tid])) {
                svals[tid] = v2;
                sidxs[tid] = i2;
            }
        }
        __syncthreads();
    }
    if (tid == 0) {
        pval[blockIdx.x] = svals[0];
        pidx[blockIdx.x] = sidxs[0];
    }
}

// ---------------------------------------------------------------------------
// Step kernel B: reduce block partials -> selected index; gather output row,
// publish next center + its norm.
// ---------------------------------------------------------------------------
__global__ __launch_bounds__(BLKT) void fps_step_select(
    const float* __restrict__ pval, const int* __restrict__ pidx, int nparts,
    const float* __restrict__ feat, const float* __restrict__ nrm,
    float* __restrict__ out, int k, float* __restrict__ center,
    float* __restrict__ cnrm)
{
    __shared__ float svals[BLKT];
    __shared__ int   sidxs[BLKT];
    int tid = threadIdx.x;

    float best = -__builtin_inff();
    int   bidx = 0x7fffffff;
    for (int i = tid; i < nparts; i += BLKT) {
        float v  = pval[i];
        int   ix = pidx[i];
        if (v > best || (v == best && ix < bidx)) { best = v; bidx = ix; }
    }
    svals[tid] = best;
    sidxs[tid] = bidx;
    __syncthreads();
    #pragma unroll
    for (int s = BLKT / 2; s > 0; s >>= 1) {
        if (tid < s) {
            float v2 = svals[tid + s];
            int   i2 = sidxs[tid + s];
            if (v2 > svals[tid] || (v2 == svals[tid] && i2 < sidxs[tid])) {
                svals[tid] = v2;
                sidxs[tid] = i2;
            }
        }
        __syncthreads();
    }
    int sel = sidxs[0];

    float v = feat[(size_t)sel * DIMS + tid];     // DIMS == blockDim.x
    out[(size_t)k * DIMS + tid] = v;
    center[tid] = v;
    if (tid == 0) *cnrm = nrm[sel];
}

// ---------------------------------------------------------------------------
extern "C" void kernel_launch(void* const* d_in, const int* in_sizes, int n_in,
                              void* d_out, int out_size, void* d_ws, size_t ws_size,
                              hipStream_t stream)
{
    (void)in_sizes; (void)n_in; (void)out_size; (void)ws_size;
    const float* feat = (const float*)d_in[0];   // [N, D] f32
    float*       out  = (float*)d_out;           // [NCORE, D] f32

    char* ws = (char*)d_ws;                       // < 1 MB total
    float* min_d  = (float*)(ws);                             // N f32
    float* nrm    = (float*)(ws + 400000);                    // N f32
    float* center = (float*)(ws + 800000);                    // DIMS f32
    float* cnrm   = (float*)(ws + 800000 + 1024);             // 1 f32
    float* pval   = (float*)(ws + 800000 + 1040);             // NBLK_A f32
    int*   pidx   = (int*)  (ws + 800000 + 1040 + 4096);      // NBLK_A i32

    fps_init_norms<<<(N_PTS + 7) / 8, BLKT, 0, stream>>>(feat, nrm, min_d);
    fps_init_seed<<<1, BLKT, 0, stream>>>(feat, nrm, out, center, cnrm);

    for (int k = 1; k < NCORE; ++k) {
        fps_step_dist<<<NBLK_A, BLKT, 0, stream>>>(
            feat, center, cnrm, nrm, min_d, pval, pidx);
        fps_step_select<<<1, BLKT, 0, stream>>>(
            pval, pidx, NBLK_A, feat, nrm, out, k, center, cnrm);
    }
}
